// RelationAwareAttentionHead_23149873725512
// MI455X (gfx1250) — matrix operand order repaired and checked by reading, not compiled, not run on hardware
//
#include <hip/hip_runtime.h>

// ---------------------------------------------------------------------------
// Relation-aware attention head (Shaw relative-position bias), MI455X / gfx1250
//
// Roofline: bias tensors (2 x 256MB fp32) dominate bytes -> bandwidth bound
// (~22us floor @ 23.3 TB/s with cross-batch L2 reuse). WMMA bf16 handles the
// real GEMMs (projections, q.kT, w.v); the per-q-row bias einsums are VALU
// streams coalesced over the contiguous d dimension. k/v tiles are staged to
// LDS with CDNA5 async global->LDS DMA (ASYNCcnt).
// ---------------------------------------------------------------------------

typedef __bf16 bf16;
typedef __attribute__((ext_vector_type(16))) __bf16 v16bf;
typedef __attribute__((ext_vector_type(8)))  float  v8f;

union AFrag { v16bf v; bf16  e[16]; };
union CFrag { v8f   v; float e[8];  };

#define NB 8      // batches
#define NS 1024   // sequence
#define NH 1024   // hidden
#define ND 64     // head dim

// A-matrix (16x32 bf16) element -> K index for lane L, vector element h.
// Contiguous K order (matches B fills): K = (lane>=16?16:0) + h
__device__ __forceinline__ int a_kidx(int L, int h) {
  return ((L & 16) ? 16 : 0) + h;
}

// CDNA5 async global->LDS: 16B per enabled lane, tracked by ASYNCcnt.
// LDS operand = low 32 bits of the generic pointer (ISA 10.2).
__device__ __forceinline__ void async_ld_b128(void* lds_dst, const void* gsrc) {
  unsigned l = (unsigned)(unsigned long long)lds_dst;
  asm volatile("global_load_async_to_lds_b128 %0, %1, off"
               :: "v"(l), "v"(gsrc) : "memory");
}
__device__ __forceinline__ void wait_async0() {
  asm volatile("s_wait_asynccnt 0x0" ::: "memory");
}

// ---------------------------------------------------------------------------
// Projection: out[m,n] = X[m,:] @ W[:,n] + b[n],  M=8192, K=1024, N=64.
// blockIdx.y selects {q,k,v}. 128 threads = 4 waves; wave w owns N-tile w.
// ---------------------------------------------------------------------------
__global__ __launch_bounds__(128)
void raah_proj_kernel(const float* __restrict__ q_in, const float* __restrict__ k_in,
                      const float* __restrict__ v_in,
                      const float* __restrict__ Wq, const float* __restrict__ Wk,
                      const float* __restrict__ Wv,
                      const float* __restrict__ bq, const float* __restrict__ bk,
                      const float* __restrict__ bv,
                      bf16* __restrict__ qbf, bf16* __restrict__ kbf,
                      bf16* __restrict__ vbf, float* __restrict__ qf32)
{
  __shared__ float Xs[16 * 32];   // X tile  [16 rows][32 K]
  __shared__ float Ws[32 * 64];   // W chunk [32 K][64 N]

  const int tid = threadIdx.x;
  const int L   = tid & 31;
  const int w   = tid >> 5;
  const int m0  = blockIdx.x * 16;
  const int p   = blockIdx.y;

  const float* X    = (p == 0) ? q_in : (p == 1) ? k_in : v_in;
  const float* W    = (p == 0) ? Wq   : (p == 1) ? Wk   : Wv;
  const float* bias = (p == 0) ? bq   : (p == 1) ? bk   : bv;
  bf16*        obf  = (p == 0) ? qbf  : (p == 1) ? kbf  : vbf;

  CFrag c;
  #pragma unroll
  for (int r = 0; r < 8; ++r) c.e[r] = 0.0f;

  for (int kc = 0; kc < NH; kc += 32) {
    // Vectorized staging: b128 loads/stores, multiple loads in flight.
    {  // Xs: 128 float4, one per thread. elem=4t: row=t>>3, col=(t&7)*4
      const float4 xv =
          ((const float4*)(X + (size_t)(m0 + (tid >> 3)) * NH + kc))[tid & 7];
      ((float4*)Xs)[tid] = xv;
    }
    {  // Ws: 512 float4, 4 per thread. c4=t+128u: row=c4>>4, col=(c4&15)*4
      float4 wv[4];
      #pragma unroll
      for (int u = 0; u < 4; ++u) {
        const int c4 = tid + 128 * u;
        wv[u] = ((const float4*)(W + (size_t)(kc + (c4 >> 4)) * ND))[c4 & 15];
      }
      #pragma unroll
      for (int u = 0; u < 4; ++u) ((float4*)Ws)[tid + 128 * u] = wv[u];
    }
    __syncthreads();

    AFrag a, bfr;
    #pragma unroll
    for (int h = 0; h < 16; ++h)
      a.e[h] = (bf16)Xs[(L & 15) * 32 + a_kidx(L, h)];
    #pragma unroll
    for (int h = 0; h < 16; ++h)   // B[K][N]: K=(L>=16?16:0)+h, N=w*16+(L&15)
      bfr.e[h] = (bf16)Ws[(((L & 16) ? 16 : 0) + h) * 64 + w * 16 + (L & 15)];

    c.v = __builtin_amdgcn_wmma_f32_16x16x32_bf16(false, a.v, false, bfr.v,
                                                  (short)0, c.v, false, false);
    __syncthreads();
  }

  #pragma unroll
  for (int r = 0; r < 8; ++r) {
    const int m = r + ((L & 16) ? 8 : 0);
    const int n = w * 16 + (L & 15);
    const float val = c.e[r] + bias[n];
    const size_t idx = (size_t)(m0 + m) * ND + n;
    obf[idx] = (bf16)val;
    if (p == 0) qf32[idx] = val;
  }
}

// ---------------------------------------------------------------------------
// Fused attention. One block = (q-tile of 16, batch). 256 threads = 8 waves.
// Grid order qt-major (qt = bx>>3, b = bx&7) so the 8 batches sharing a
// 4MB k_bias/v_bias slice run adjacently -> L2 reuse, ~512MB net HBM traffic.
// ---------------------------------------------------------------------------
__global__ __launch_bounds__(256)
void raah_attn_kernel(const bf16* __restrict__ qbf, const bf16* __restrict__ kbf,
                      const bf16* __restrict__ vbf, const float* __restrict__ qf32,
                      const float* __restrict__ kbias, const float* __restrict__ vbias,
                      const int* __restrict__ mask, float* __restrict__ out)
{
  extern __shared__ char smem[];
  float* s_scores = (float*)smem;                 // [16][1024] f32 (64KB)
  float* s_qf     = s_scores + 16 * 1024;         // [16][64]  f32 (4KB)
  float* s_oacc   = s_qf + 16 * 64;               // [16][64]  f32 (4KB)
  bf16*  s_wbf    = (bf16*)(s_oacc + 16 * 64);    // [16][1024] bf16 (32KB)
  bf16*  s_qbf    = s_wbf + 16 * 1024;            // [16][64]  bf16 (2KB)
  bf16*  s_stage  = s_qbf + 16 * 64;              // 8 waves x [32][64] bf16 (32KB)

  const int tid = threadIdx.x;
  const int L   = tid & 31;
  const int w   = tid >> 5;
  const int qt  = blockIdx.x >> 3;
  const int b   = blockIdx.x & 7;
  const int q0  = qt * 16;

  // Stage q tile (vectorized) and zero the output accumulator.
  {
    const size_t gq = (size_t)(b * NS + q0) * ND;
    ((float4*)s_qf)[tid]   = ((const float4*)(qf32 + gq))[tid];   // 256 x 16B
    ((float4*)s_oacc)[tid] = float4{0.f, 0.f, 0.f, 0.f};          // 256 x 16B, one per thread
    if (tid < 128)
      ((uint4*)s_qbf)[tid] = ((const uint4*)(qbf + gq))[tid];     // 128 x 16B
  }
  __syncthreads();

  bf16* st = s_stage + w * (32 * 64);

  // ---- Phase A: scores = (q.kT + q.k_bias) / sqrt(D), masked -> LDS -------
  // Each wave owns k-tiles kt = w, w+8, ... (uniform 8 iterations per wave).
  for (int kt = w; kt < 64; kt += 8) {
    // Stage k tile [16 rows][64 d] bf16 via async DMA: 2KB = 4 x (32 lanes x 16B)
    #pragma unroll
    for (int u = 0; u < 4; ++u) {
      const int c = L + 32 * u;            // 16B chunk id; halves 8c..8c+7
      async_ld_b128(st + 8 * c,
                    kbf + (size_t)(b * NS + kt * 16 + (c >> 3)) * ND + (c & 7) * 8);
    }
    wait_async0();

    // attn1 via WMMA: C[i,j] = sum_d q[i,d] * k[j,d], K=64 in two x32 steps
    CFrag c;
    #pragma unroll
    for (int r = 0; r < 8; ++r) c.e[r] = 0.0f;
    #pragma unroll
    for (int db = 0; db < 64; db += 32) {
      AFrag a, bb;
      #pragma unroll
      for (int h = 0; h < 16; ++h)
        a.e[h] = s_qbf[(L & 15) * 64 + db + a_kidx(L, h)];
      #pragma unroll
      for (int h = 0; h < 16; ++h)   // B[K=d][N=j] = k[j][d]
        bb.e[h] = st[(L & 15) * 64 + db + ((L & 16) ? 16 : 0) + h];
      c.v = __builtin_amdgcn_wmma_f32_16x16x32_bf16(false, a.v, false, bb.v,
                                                    (short)0, c.v, false, false);
    }

    // attn2 (VALU stream): lane = (j = L&15, d-half = L>>4); shfl_xor(16) joins
    const int j  = L & 15;
    const int dh = (L >> 4) & 1;
    for (int i = 0; i < 16; ++i) {
      const float* kbp = kbias + ((size_t)(q0 + i) * NS + kt * 16 + j) * ND + dh * 32;
      if (i < 15) __builtin_prefetch(kbp + (size_t)NS * ND, 0, 1);  // next q row
      float p2 = 0.0f;
      #pragma unroll 8
      for (int dd = 0; dd < 32; ++dd)
        p2 += s_qf[i * 64 + dh * 32 + dd] * kbp[dd];
      p2 += __shfl_xor(p2, 16, 32);
      if (L < 16) s_scores[i * 1024 + kt * 16 + j] = p2;   // attn2 staged
    }

    // combine + scale + mask (each (m,col) touched by exactly one lane)
    const int mval = mask[b * NS + kt * 16 + (L & 15)];
    #pragma unroll
    for (int r = 0; r < 8; ++r) {
      const int m   = r + ((L & 16) ? 8 : 0);
      const int col = kt * 16 + (L & 15);
      float sc = (s_scores[m * 1024 + col] + c.e[r]) * 0.125f;  // 1/sqrt(64)
      if (mval == 0) sc = -1e9f;
      s_scores[m * 1024 + col] = sc;
    }
    __syncthreads();   // uniform: every wave runs 8 iterations
  }

  // ---- Phase B: row softmax, wave w handles rows 2w, 2w+1 -----------------
  for (int ri = 0; ri < 2; ++ri) {
    const int i = w * 2 + ri;
    float mx = -3.4e38f;
    for (int c2 = L; c2 < 1024; c2 += 32)
      mx = fmaxf(mx, s_scores[i * 1024 + c2]);
    #pragma unroll
    for (int mb = 16; mb >= 1; mb >>= 1)
      mx = fmaxf(mx, __shfl_xor(mx, mb, 32));
    float sm = 0.0f;
    for (int c2 = L; c2 < 1024; c2 += 32) {
      const float ev = __expf(s_scores[i * 1024 + c2] - mx);
      s_scores[i * 1024 + c2] = ev;
      sm += ev;
    }
    #pragma unroll
    for (int mb = 16; mb >= 1; mb >>= 1)
      sm += __shfl_xor(sm, mb, 32);
    const float inv = 1.0f / sm;
    for (int c2 = L; c2 < 1024; c2 += 32) {
      const float wv = s_scores[i * 1024 + c2] * inv;
      s_scores[i * 1024 + c2] = wv;          // f32 weights (for values_2)
      s_wbf[i * 1024 + c2] = (bf16)wv;       // bf16 weights (for WMMA)
    }
  }
  __syncthreads();

  // ---- Phase C1: values_1 = w @ v via WMMA, K split across waves ----------
  CFrag acc[4];
  #pragma unroll
  for (int nt = 0; nt < 4; ++nt)
    #pragma unroll
    for (int r = 0; r < 8; ++r) acc[nt].e[r] = 0.0f;

  for (int kc = w; kc < 32; kc += 8) {       // K chunks of 32, 4 per wave
    // Stage v chunk [32 j][64 d] bf16 via async DMA: 4KB = 8 x (32 x 16B)
    #pragma unroll
    for (int u = 0; u < 8; ++u) {
      const int c = L + 32 * u;
      async_ld_b128(st + 8 * c,
                    vbf + (size_t)(b * NS + kc * 32 + (c >> 3)) * ND + (c & 7) * 8);
    }
    wait_async0();

    AFrag a;                                  // A[i][jj] = w[i, kc*32+jj]
    #pragma unroll
    for (int h = 0; h < 16; ++h)
      a.e[h] = s_wbf[(L & 15) * 1024 + kc * 32 + a_kidx(L, h)];
    #pragma unroll
    for (int nt = 0; nt < 4; ++nt) {
      AFrag bb;                               // B[jj][n] = v[jj][nt*16+n]
      #pragma unroll
      for (int h = 0; h < 16; ++h)
        bb.e[h] = st[(((L & 16) ? 16 : 0) + h) * 64 + nt * 16 + (L & 15)];
      acc[nt].v = __builtin_amdgcn_wmma_f32_16x16x32_bf16(
          false, a.v, false, bb.v, (short)0, acc[nt].v, false, false);
    }
  }
  #pragma unroll
  for (int nt = 0; nt < 4; ++nt)
    #pragma unroll
    for (int r = 0; r < 8; ++r)   // cross-wave K reduction: ds_add_f32
      atomicAdd(&s_oacc[(r + ((L & 16) ? 8 : 0)) * 64 + nt * 16 + (L & 15)],
                acc[nt].e[r]);
  __syncthreads();

  // ---- Phase C2: values_2 (VALU stream over v_bias, coalesced over d) -----
  const int d  = tid & 63;
  const int ib = tid >> 6;
  for (int i = ib; i < 16; i += 4) {
    const float* vbp = vbias + (size_t)(q0 + i) * NS * ND + d;
    float a2 = 0.0f;
    for (int j0 = 0; j0 < 1024; j0 += 8) {
      if ((j0 & 31) == 0 && j0 < 960)
        __builtin_prefetch(vbp + (size_t)(j0 + 64) * ND, 0, 1);
      #pragma unroll
      for (int u = 0; u < 8; ++u)
        a2 += s_scores[i * 1024 + j0 + u] * vbp[(size_t)(j0 + u) * ND];
    }
    out[(size_t)(b * NS + q0 + i) * ND + d] = a2 + s_oacc[i * 64 + d];
  }
}

// ---------------------------------------------------------------------------
extern "C" void kernel_launch(void* const* d_in, const int* in_sizes, int n_in,
                              void* d_out, int out_size, void* d_ws, size_t ws_size,
                              hipStream_t stream) {
  const float* query = (const float*)d_in[0];
  const float* key   = (const float*)d_in[1];
  const float* value = (const float*)d_in[2];
  const float* Wq    = (const float*)d_in[3];
  const float* bq    = (const float*)d_in[4];
  const float* Wk    = (const float*)d_in[5];
  const float* bk    = (const float*)d_in[6];
  const float* Wv    = (const float*)d_in[7];
  const float* bv    = (const float*)d_in[8];
  const float* kbias = (const float*)d_in[9];
  const float* vbias = (const float*)d_in[10];
  const int*   mask  = (const int*)d_in[11];

  // Workspace: q/k/v bf16 (1MB each) + q f32 (2MB) = 5MB
  char*  ws   = (char*)d_ws;
  bf16*  qbf  = (bf16*)(ws);
  bf16*  kbf  = (bf16*)(ws + (1u << 20));
  bf16*  vbf  = (bf16*)(ws + 2u * (1u << 20));
  float* qf32 = (float*)(ws + 3u * (1u << 20));

  raah_proj_kernel<<<dim3(512, 3), 128, 0, stream>>>(
      query, key, value, Wq, Wk, Wv, bq, bk, bv, qbf, kbf, vbf, qf32);

  const size_t smem = 16 * 1024 * 4   // s_scores
                    + 16 * 64 * 4     // s_qf
                    + 16 * 64 * 4     // s_oacc
                    + 16 * 1024 * 2   // s_wbf
                    + 16 * 64 * 2     // s_qbf
                    + 8 * 32 * 64 * 2;// s_stage  (= 141312 B < 160KB)
  hipFuncSetAttribute((const void*)raah_attn_kernel,
                      hipFuncAttributeMaxDynamicSharedMemorySize, (int)smem);

  raah_attn_kernel<<<dim3(NB * (NS / 16)), 256, smem, stream>>>(
      qbf, kbf, vbf, qf32, kbias, vbias, mask, (float*)d_out);
}
